// GNN_48515950575687
// MI455X (gfx1250) — matrix operand order repaired and checked by reading, not compiled, not run on hardware
//
#include <hip/hip_runtime.h>

typedef __attribute__((ext_vector_type(16))) _Float16 v16h;
typedef __attribute__((ext_vector_type(8)))  float    v8f;

union AFrag { v16h h; unsigned int u[8]; };

#define TSTRIDE 132   // per-thread LDS region stride in dwords (16B aligned, bank-staggered)

__device__ __forceinline__ unsigned int pack_h2(float a, float b) {
    union { _Float16 h[2]; unsigned int u; } p;
    p.h[0] = (_Float16)a; p.h[1] = (_Float16)b;
    return p.u;
}

// ---------------------------------------------------------------------------
// Kernel 0: convert Linear weights to f16 in workspace (row-major, so that
// (k, k+1) half-pairs are dword-contiguous for WMMA operand builds).
// ---------------------------------------------------------------------------
__global__ void gnn_convert_weights(const float* __restrict__ wl1,
                                    const float* __restrict__ wl2,
                                    _Float16* __restrict__ wl1h,
                                    _Float16* __restrict__ wl2h) {
    int i = blockIdx.x * blockDim.x + threadIdx.x;
    if (i < 64 * 128) wl1h[i] = (_Float16)wl1[i];
    if (i < 16 * 64)  wl2h[i] = (_Float16)wl2[i];
}

// ---------------------------------------------------------------------------
// One GCN conv layer, per-lane private LDS regions, float4 DS traffic.
// dst[n] is seeded with the self-loop message (norm = 1/deg), then only the
// 16 real edges are scatter-accumulated.
// ---------------------------------------------------------------------------
template<int CIN, int COUT>
__device__ __forceinline__ void gcn_layer(float* __restrict__ dstL,
                                          float* __restrict__ msgL,
                                          const float* hin,      // [8*CIN] regs
                                          const float* sWW,      // [COUT][CIN]
                                          const float* norm,     // [24]
                                          const int* rowe, const int* cole)
{
    // messages hw[n][*] -> msgL ; self-loop init -> dstL
    #pragma unroll
    for (int n = 0; n < 8; n++) {
        const float ns = norm[16 + n];
        #pragma unroll
        for (int c0 = 0; c0 < COUT; c0 += 4) {
            float hw[4];
            #pragma unroll
            for (int j = 0; j < 4; j++) {
                float acc = 0.0f;
                #pragma unroll
                for (int f = 0; f < CIN; f++)
                    acc += hin[n * CIN + f] * sWW[(c0 + j) * CIN + f];
                hw[j] = acc;
            }
            float4 m; m.x = hw[0]; m.y = hw[1]; m.z = hw[2]; m.w = hw[3];
            *(float4*)&msgL[n * COUT + c0] = m;
            float4 s; s.x = ns * hw[0]; s.y = ns * hw[1];
            s.z = ns * hw[2]; s.w = ns * hw[3];
            *(float4*)&dstL[n * COUT + c0] = s;
        }
    }
    // scatter the 16 real edges (read-modify-write, per-thread private region)
    #pragma unroll
    for (int e = 0; e < 16; e++) {
        const int   r  = rowe[e];
        const int   cl = cole[e];
        const float nm = norm[e];
        #pragma unroll
        for (int c0 = 0; c0 < COUT; c0 += 4) {
            float4 m = *(const float4*)&msgL[r  * COUT + c0];
            float4 d = *(float4*)      &dstL[cl * COUT + c0];
            d.x += nm * m.x; d.y += nm * m.y;
            d.z += nm * m.z; d.w += nm * m.w;
            *(float4*)&dstL[cl * COUT + c0] = d;
        }
    }
}

template<int C>
__device__ __forceinline__ void gcn_relu_read(const float* __restrict__ dstL,
                                              const float* bias, float* hout)
{
    #pragma unroll
    for (int n = 0; n < 8; n++)
        #pragma unroll
        for (int c0 = 0; c0 < C; c0 += 4) {
            float4 v = *(const float4*)&dstL[n * C + c0];
            hout[n * C + c0 + 0] = fmaxf(v.x + bias[c0 + 0], 0.0f);
            hout[n * C + c0 + 1] = fmaxf(v.y + bias[c0 + 1], 0.0f);
            hout[n * C + c0 + 2] = fmaxf(v.z + bias[c0 + 2], 0.0f);
            hout[n * C + c0 + 3] = fmaxf(v.w + bias[c0 + 3], 0.0f);
        }
}

// ---------------------------------------------------------------------------
// Fused GCN (per-lane, LDS-indexed) + WMMA MLP head.
// 1 wave per block, 1 graph per lane, 32 graphs per block.
// ---------------------------------------------------------------------------
__global__ __launch_bounds__(32)
void gnn_fused(const float* __restrict__ x,
               const int*   __restrict__ ei,
               const float* __restrict__ W1, const float* __restrict__ b1,
               const float* __restrict__ W2, const float* __restrict__ b2,
               const float* __restrict__ W3, const float* __restrict__ b3,
               const float* __restrict__ bl1, const float* __restrict__ bl2,
               const _Float16* __restrict__ wl1h,
               const _Float16* __restrict__ wl2h,
               float* __restrict__ out)
{
    __shared__ float sA[TSTRIDE * 32];   // per-thread dst regions / z-stage
    __shared__ float sB[TSTRIDE * 32];   // per-thread msg regions / f16 A-stage
    __shared__ float sW[256];            // staged conv weights + biases

    const int  lane = threadIdx.x;
    const long gblk = (long)blockIdx.x * 32;
    const long g    = gblk + lane;

    float* dstL = sA + lane * TSTRIDE;   // 128 dwords used, 16B-aligned base
    float* msgL = sB + lane * TSTRIDE;

    // ---- stage conv weights: [0..7]=W1, [8..11]=b1, [12..43]=W2,
    //      [44..51]=b2, [52..179]=W3, [180..195]=b3
    if (lane < 8)  sW[lane]      = W1[lane];
    if (lane < 4)  sW[8 + lane]  = b1[lane];
    sW[12 + lane] = W2[lane];                       // 32 elements, one per lane
    if (lane < 8)  sW[44 + lane] = b2[lane];
    #pragma unroll
    for (int i = lane; i < 128; i += 32) sW[52 + i] = W3[i];
    if (lane < 16) sW[180 + lane] = b3[lane];
    __syncthreads();

    // ---- per-graph inputs
    float xv[16];
    const float* xp = x + g * 16;
    #pragma unroll
    for (int i = 0; i < 16; i += 4) {
        float4 v = *(const float4*)&xp[i];
        xv[i] = v.x; xv[i + 1] = v.y; xv[i + 2] = v.z; xv[i + 3] = v.w;
    }

    int rowe[16], cole[16];
    const int* ep = ei + g * 32;
    #pragma unroll
    for (int e = 0; e < 16; e++) { rowe[e] = ep[e]; cole[e] = ep[16 + e]; }

    // ---- degrees: packed 8-bit counters (max degree 17 < 256, no overflow).
    // Self-loop contributes 1 to every node: start every byte at 1.
    unsigned long long degp = 0x0101010101010101ull;
    #pragma unroll
    for (int e = 0; e < 16; e++)
        degp += 1ull << (cole[e] * 8);
    float deg[8];
    #pragma unroll
    for (int n = 0; n < 8; n++)
        deg[n] = (float)((unsigned)(degp >> (n * 8)) & 0xFFu);

    // dis into private LDS cells for dynamic-index gather
    #pragma unroll
    for (int n = 0; n < 8; n++) dstL[n] = rsqrtf(deg[n]);

    float norm[24];
    #pragma unroll
    for (int e = 0; e < 16; e++) norm[e] = dstL[rowe[e]] * dstL[cole[e]];
    #pragma unroll
    for (int n = 0; n < 8; n++)  norm[16 + n] = 1.0f / deg[n];   // dis[n]^2

    // ---- three GCN layers
    gcn_layer<2, 4>(dstL, msgL, xv, sW + 0, norm, rowe, cole);
    float h1[32];
    gcn_relu_read<4>(dstL, sW + 8, h1);

    gcn_layer<4, 8>(dstL, msgL, h1, sW + 12, norm, rowe, cole);
    float h2[64];
    gcn_relu_read<8>(dstL, sW + 44, h2);

    gcn_layer<8, 16>(dstL, msgL, h2, sW + 52, norm, rowe, cole);
    __syncthreads();

    // ---- h3 = relu(dst + b3) -> f16 A-stage aw[32 graphs][128] over sB
    _Float16* aw = (_Float16*)sB;
    #pragma unroll
    for (int k0 = 0; k0 < 128; k0 += 8) {
        float4 v0 = *(const float4*)&dstL[k0];
        float4 v1 = *(const float4*)&dstL[k0 + 4];
        const float* bb = &sW[180 + (k0 & 15)];
        float r0 = fmaxf(v0.x + bb[0], 0.0f);
        float r1 = fmaxf(v0.y + bb[1], 0.0f);
        float r2 = fmaxf(v0.z + bb[2], 0.0f);
        float r3 = fmaxf(v0.w + bb[3], 0.0f);
        float r4 = fmaxf(v1.x + bb[4], 0.0f);
        float r5 = fmaxf(v1.y + bb[5], 0.0f);
        float r6 = fmaxf(v1.z + bb[6], 0.0f);
        float r7 = fmaxf(v1.w + bb[7], 0.0f);
        uint4 uu;
        uu.x = pack_h2(r0, r1); uu.y = pack_h2(r2, r3);
        uu.z = pack_h2(r4, r5); uu.w = pack_h2(r6, r7);
        *(uint4*)&aw[lane * 128 + k0] = uu;
    }
    __syncthreads();

    // =========================== WMMA MLP head ==============================
    _Float16* zs = (_Float16*)sA;        // z-stage: [32 graphs][64] f16
    const int ln15 = lane & 15;
    const int mofs = (lane & 16) >> 1;   // +8 row offset for hi half (C layout)
    const int aofs = (lane & 16) >> 1;   // +8  K offset for hi half (A frags)
    const int bofs = (lane & 16);        // +16 for hi half (unused; B frags use aofs, +8 K)

    #pragma unroll
    for (int t = 0; t < 2; t++) {        // two 16-graph M-tiles
        // A fragments for all 4 K-chunks (K=128)
        AFrag a[4];
        #pragma unroll
        for (int kc = 0; kc < 4; kc++)
            #pragma unroll
            for (int v = 0; v < 8; v++) {
                int kb = kc * 32 + ((v < 4) ? 0 : 16) + aofs + 2 * (v & 3);
                a[kc].u[v] = *(const unsigned int*)&aw[(t * 16 + ln15) * 128 + kb];
            }
        // Linear1: z[16x64] = A[16x128] * Wl1^T + bl1, ReLU
        #pragma unroll
        for (int nt = 0; nt < 4; nt++) {
            float bias = bl1[nt * 16 + ln15];
            v8f c;
            #pragma unroll
            for (int i = 0; i < 8; i++) c[i] = bias;
            #pragma unroll
            for (int kc = 0; kc < 4; kc++) {
                AFrag bf;
                #pragma unroll
                for (int v = 0; v < 8; v++) {
                    int kb = kc * 32 + ((v < 4) ? 0 : 16) + aofs + 2 * (v & 3);
                    bf.u[v] = *(const unsigned int*)&wl1h[(nt * 16 + ln15) * 128 + kb];
                }
                c = __builtin_amdgcn_wmma_f32_16x16x32_f16(
                        false, a[kc].h, false, bf.h, (short)0, c, false, false);
            }
            #pragma unroll
            for (int r = 0; r < 8; r++) {
                int m = r + mofs;
                zs[(t * 16 + m) * 64 + nt * 16 + ln15] =
                    (_Float16)fmaxf(c[r], 0.0f);
            }
        }
        __syncthreads();

        // Linear2: out[16x16] = z[16x64] * Wl2^T + bl2
        float bias2 = bl2[ln15];
        v8f c2;
        #pragma unroll
        for (int i = 0; i < 8; i++) c2[i] = bias2;
        #pragma unroll
        for (int kc = 0; kc < 2; kc++) {
            AFrag a2, b2;
            #pragma unroll
            for (int v = 0; v < 8; v++) {
                int kba = kc * 32 + ((v < 4) ? 0 : 16) + aofs + 2 * (v & 3);
                a2.u[v] = *(const unsigned int*)&zs[(t * 16 + ln15) * 64 + kba];
                int kbb = kc * 32 + ((v < 4) ? 0 : 16) + aofs + 2 * (v & 3);
                b2.u[v] = *(const unsigned int*)&wl2h[ln15 * 64 + kbb];
            }
            c2 = __builtin_amdgcn_wmma_f32_16x16x32_f16(
                     false, a2.h, false, b2.h, (short)0, c2, false, false);
        }
        #pragma unroll
        for (int r = 0; r < 8; r++) {
            int m = r + mofs;
            out[(gblk + t * 16 + m) * 16 + ln15] = c2[r];
        }
        __syncthreads();
    }
}

// ---------------------------------------------------------------------------
extern "C" void kernel_launch(void* const* d_in, const int* in_sizes, int n_in,
                              void* d_out, int out_size, void* d_ws, size_t ws_size,
                              hipStream_t stream) {
    const float* x   = (const float*)d_in[0];
    const int*   ei  = (const int*)  d_in[1];
    const float* W1  = (const float*)d_in[2];
    const float* b1  = (const float*)d_in[3];
    const float* W2  = (const float*)d_in[4];
    const float* b2  = (const float*)d_in[5];
    const float* W3  = (const float*)d_in[6];
    const float* b3  = (const float*)d_in[7];
    const float* Wl1 = (const float*)d_in[8];
    const float* bl1 = (const float*)d_in[9];
    const float* Wl2 = (const float*)d_in[10];
    const float* bl2 = (const float*)d_in[11];
    float* out = (float*)d_out;

    const int B = in_sizes[0] / 16;            // x is [B, 8, 2]

    _Float16* wl1h = (_Float16*)d_ws;          // 64*128 halves
    _Float16* wl2h = wl1h + 64 * 128;          // 16*64 halves

    gnn_convert_weights<<<(64 * 128 + 255) / 256, 256, 0, stream>>>(
        Wl1, Wl2, wl1h, wl2h);

    gnn_fused<<<B / 32, 32, 0, stream>>>(
        x, ei, W1, b1, W2, b2, W3, b3, bl1, bl2, wl1h, wl2h, out);
}
